// MSTDependencyParser_54589034332326
// MI455X (gfx1250) — compile-verified, matrix operand order checked by reading, not run on hardware
//
#include <hip/hip_runtime.h>
#include <hip/hip_bf16.h>
#include <stdint.h>

// ---------------------------------------------------------------------------
// Types for CDNA5 WMMA (wave32): v_wmma_f32_16x16x32_f16
// ---------------------------------------------------------------------------
typedef __attribute__((ext_vector_type(16))) _Float16 v16h;
typedef __attribute__((ext_vector_type(8)))  _Float16 v8h;
typedef __attribute__((ext_vector_type(8)))  float    v8f;

__device__ __forceinline__ float fast_tanh(float x) {
#if __has_builtin(__builtin_amdgcn_tanhf)
  return __builtin_amdgcn_tanhf(x);   // gfx1250 v_tanh_f32 (TRANS unit)
#else
  return tanhf(x);
#endif
}
__device__ __forceinline__ float fast_sigmoid(float x) {
  return 1.0f / (1.0f + __expf(-x));
}

// ---------------------------------------------------------------------------
// Fragment load: 16-bit operand, 16x32 (A) or 32x16-as-NxK (B given row-major
// [N][K] = B^T).  Per ISA 7.12.2: lane<16 holds row (lane&15), K {0..7} then
// {16..23}; lane>=16 holds K {8..15} then {24..31}.  Two b128 loads per lane.
// base must point at row (lane&15) + k0 offset already applied by caller.
// ---------------------------------------------------------------------------
__device__ __forceinline__ v16h load_frag16(const _Float16* p) {
  v8h lo = *(const v8h*)(p);
  v8h hi = *(const v8h*)(p + 16);
  v16h a;
#pragma unroll
  for (int t = 0; t < 8; ++t) { a[t] = lo[t]; a[8 + t] = hi[t]; }
  return a;
}

// ---------------------------------------------------------------------------
// GEMM: C[M][ldc](f32) = A[M][K](f16,row-major) * Bt[N][K](f16,row-major)^T
//       + bias[n] (optional, padded to N).
// K % 32 == 0, N % 16 == 0, M % 16 == 0.  One 16x16 tile per wave, 4 waves
// per block -> 64 output columns per block.
// ---------------------------------------------------------------------------
__global__ __launch_bounds__(128)
void wmma_gemm_nt(const _Float16* __restrict__ A, const _Float16* __restrict__ Bt,
                  const float* __restrict__ bias, float* __restrict__ C,
                  int M, int Nc, int K, int ldc) {
  const int lane = threadIdx.x & 31;
  const int wave = threadIdx.x >> 5;
  const int m0 = blockIdx.x * 16;
  const int n0 = (blockIdx.y * 4 + wave) * 16;
  const int r  = lane & 15;
  const int k0 = (lane >> 4) * 8;

  const _Float16* pa = A  + (size_t)(m0 + r) * K + k0;
  const _Float16* pb = Bt + (size_t)(n0 + r) * K + k0;

  v8f acc = {0.f, 0.f, 0.f, 0.f, 0.f, 0.f, 0.f, 0.f};
  for (int kk = 0; kk < K; kk += 32) {
    v16h a = load_frag16(pa + kk);
    v16h b = load_frag16(pb + kk);
    acc = __builtin_amdgcn_wmma_f32_16x16x32_f16(
        /*neg_a=*/false, a, /*neg_b=*/false, b,
        /*c_mod=*/(short)0, acc, /*reuse_a=*/false, /*reuse_b=*/false);
  }

  // D layout: lane<16 -> rows 0..7 col=lane ; lane>=16 -> rows 8..15 col=lane-16
  const int col   = lane & 15;
  const int rbase = (lane >> 4) * 8;
  const float bv  = bias ? bias[n0 + col] : 0.0f;
  float* out = C + (size_t)(m0 + rbase) * ldc + n0 + col;
#pragma unroll
  for (int v = 0; v < 8; ++v) out[(size_t)v * ldc] = acc[v] + bv;
}

// ---------------------------------------------------------------------------
// Prep / marshalling kernels
// ---------------------------------------------------------------------------
__global__ void pack_f16_k(const float* __restrict__ W, _Float16* __restrict__ out,
                           int rows, int cols, int ld, int coloff,
                           int orows, int ocols) {
  int total = orows * ocols;
  for (int idx = blockIdx.x * blockDim.x + threadIdx.x; idx < total;
       idx += gridDim.x * blockDim.x) {
    int rr = idx / ocols, cc = idx % ocols;
    float v = (rr < rows && cc < cols) ? W[(size_t)rr * ld + coloff + cc] : 0.f;
    out[idx] = (_Float16)v;
  }
}

__global__ void pack_f32_k(const float* __restrict__ W, float* __restrict__ out,
                           int rows, int cols, int ld, int coloff,
                           int orows, int ocols) {
  int total = orows * ocols;
  for (int idx = blockIdx.x * blockDim.x + threadIdx.x; idx < total;
       idx += gridDim.x * blockDim.x) {
    int rr = idx / ocols, cc = idx % ocols;
    out[idx] = (rr < rows && cc < cols) ? W[(size_t)rr * ld + coloff + cc] : 0.f;
  }
}

// WhhT[128][512] <- Whh[500][125] transposed+padded (coalesced recurrence reads)
__global__ void pack_whhT_k(const float* __restrict__ Whh, float* __restrict__ out) {
  int total = 128 * 512;
  for (int idx = blockIdx.x * blockDim.x + threadIdx.x; idx < total;
       idx += gridDim.x * blockDim.x) {
    int k = idx / 512, u = idx % 512;
    out[idx] = (u < 500 && k < 125) ? Whh[(size_t)u * 125 + k] : 0.f;
  }
}

// x[1024][352] f16 = [word_vecs | pos_emb[pos_idx] | 0-pad]
__global__ void build_x_k(const float* __restrict__ wv, const int* __restrict__ pos_idx,
                          const float* __restrict__ pos_emb, _Float16* __restrict__ x) {
  int row = blockIdx.x;
  int c = threadIdx.x;                 // blockDim.x == 352
  float v = 0.f;
  if (c < 300)      v = wv[(size_t)row * 300 + c];
  else if (c < 325) v = pos_emb[(size_t)pos_idx[row] * 25 + (c - 300)];
  x[(size_t)row * 352 + c] = (_Float16)v;
}

__global__ void zero_u32_k(uint32_t* p, int n) {
  for (int i = blockIdx.x * blockDim.x + threadIdx.x; i < n;
       i += gridDim.x * blockDim.x) p[i] = 0u;
}

// ---------------------------------------------------------------------------
// BiLSTM recurrence (one block per direction).  Gates u<500; cells u<125.
// Gpre[t][512] = x@Wih^T + b already computed by WMMA GEMM.
// ---------------------------------------------------------------------------
__global__ __launch_bounds__(512)
void bilstm_recur_k(const float* __restrict__ GpreF, const float* __restrict__ GpreB,
                    const float* __restrict__ WhhTF, const float* __restrict__ WhhTB,
                    _Float16* __restrict__ hcat, int N) {
  __shared__ float h_s[125];
  __shared__ float g_s[500];
  const int dir = blockIdx.x;
  const float* Gpre = dir ? GpreB : GpreF;
  const float* WhhT = dir ? WhhTB : WhhTF;
  const int diroff = dir * 125;
  const int u = threadIdx.x;

  float c = 0.f;
  if (u < 125) h_s[u] = 0.f;
  __syncthreads();

  for (int step = 0; step < N; ++step) {
    const int t = dir ? (N - 1 - step) : step;
    if (u < 500) {
      float acc = Gpre[(size_t)t * 512 + u];
#pragma unroll 5
      for (int k = 0; k < 125; ++k)
        acc = fmaf(WhhT[k * 512 + u], h_s[k], acc);
      g_s[u] = acc;
    }
    __syncthreads();
    if (u < 125) {
      float ig = fast_sigmoid(g_s[u]);
      float fg = fast_sigmoid(g_s[125 + u]);
      float gg = fast_tanh(g_s[250 + u]);
      float og = fast_sigmoid(g_s[375 + u]);
      c = fg * c + ig * gg;
      float h = og * fast_tanh(c);
      h_s[u] = h;
      hcat[(size_t)t * 256 + diroff + u] = (_Float16)h;
    }
    __syncthreads();
  }
}

// ---------------------------------------------------------------------------
// Pairwise scorer: S[i][j] = mask * (sum_k tanh(A[i,k]+B[j,k]+b1[k])*w2[k] + b2)
// 32x32 tile per block; A/B rows staged in LDS (pad 101 -> conflict-free).
// ---------------------------------------------------------------------------
__global__ __launch_bounds__(256)
void pair_scores_k(const float* __restrict__ Ap, const float* __restrict__ Bp,
                   const float* __restrict__ fc1_b, const float* __restrict__ fc2_W,
                   const float* __restrict__ fc2_b, float* __restrict__ S, int N) {
  __shared__ float As[32][101];
  __shared__ float Bs[32][101];
  __shared__ float w2[100];
  __shared__ float bb[100];
  const int i0 = blockIdx.x * 32, j0 = blockIdx.y * 32;
  const int tid = threadIdx.x;

  for (int e = tid; e < 32 * 100; e += 256) {
    int rr = e / 100, kk = e % 100;
    As[rr][kk] = Ap[(size_t)(i0 + rr) * 128 + kk];
    Bs[rr][kk] = Bp[(size_t)(j0 + rr) * 128 + kk];
  }
  if (tid < 100) { w2[tid] = fc2_W[tid]; bb[tid] = fc1_b[tid]; }
  __syncthreads();

  const int j  = tid & 31;
  const int ib = tid >> 5;           // 0..7 ; thread covers i = ib, ib+8, ib+16, ib+24
  const float b2 = fc2_b[0];
  float acc[4] = {0.f, 0.f, 0.f, 0.f};

  for (int k = 0; k < 100; ++k) {
    const float bv = Bs[j][k] + bb[k];
    const float wk = w2[k];
#pragma unroll
    for (int r = 0; r < 4; ++r)
      acc[r] = fmaf(fast_tanh(As[ib + r * 8][k] + bv), wk, acc[r]);
  }

  const int gj = j0 + j;
#pragma unroll
  for (int r = 0; r < 4; ++r) {
    const int gi = i0 + ib + r * 8;
    const float s = (gi == gj || gj == 0) ? 0.f : (acc[r] + b2);
    S[(size_t)gi * N + gj] = s;
  }
}

// ---------------------------------------------------------------------------
// Per-column (child j) logsumexp + NLL contribution:  m + log(sum) - S[p_j][j]
// ---------------------------------------------------------------------------
__global__ __launch_bounds__(256)
void col_loss_k(const float* __restrict__ S, const int* __restrict__ parents,
                float* __restrict__ colloss, int N) {
  __shared__ float red[256];
  const int j = blockIdx.x;
  const int tid = threadIdx.x;

  float m = -1e30f;
  for (int i = tid; i < N; i += 256) m = fmaxf(m, S[(size_t)i * N + j]);
  red[tid] = m; __syncthreads();
  for (int s = 128; s > 0; s >>= 1) {
    if (tid < s) red[tid] = fmaxf(red[tid], red[tid + s]);
    __syncthreads();
  }
  m = red[0]; __syncthreads();

  float sum = 0.f;
  for (int i = tid; i < N; i += 256) sum += __expf(S[(size_t)i * N + j] - m);
  red[tid] = sum; __syncthreads();
  for (int s = 128; s > 0; s >>= 1) {
    if (tid < s) red[tid] += red[tid + s];
    __syncthreads();
  }
  if (tid == 0) {
    const int p = parents[j];
    colloss[j] = m + __logf(red[0]) - S[(size_t)p * N + j];
  }
}

__global__ __launch_bounds__(256)
void reduce_loss_k(const float* __restrict__ colloss, float* __restrict__ out, int N) {
  __shared__ float red[256];
  const int tid = threadIdx.x;
  float s = 0.f;
  for (int i = tid; i < N; i += 256) s += colloss[i];
  red[tid] = s; __syncthreads();
  for (int k = 128; k > 0; k >>= 1) {
    if (tid < k) red[tid] += red[tid + k];
    __syncthreads();
  }
  if (tid == 0) out[0] = red[0] / (float)N;
}

// ---------------------------------------------------------------------------
// Host launcher
// ---------------------------------------------------------------------------
extern "C" void kernel_launch(void* const* d_in, const int* in_sizes, int n_in,
                              void* d_out, int out_size, void* d_ws, size_t ws_size,
                              hipStream_t stream) {
  (void)in_sizes; (void)n_in; (void)out_size; (void)ws_size;
  const int N = 1024;

  const float* word_vecs = (const float*)d_in[0];
  const int*   pos_idx   = (const int*)  d_in[1];
  const int*   parents   = (const int*)  d_in[2];
  const float* pos_emb   = (const float*)d_in[3];
  const float* Wih0      = (const float*)d_in[4];   // [2][500][325]
  const float* Whh0      = (const float*)d_in[5];   // [2][500][125]
  const float* b0        = (const float*)d_in[6];   // [2][500]
  const float* Wih1      = (const float*)d_in[7];   // [2][500][250]
  const float* Whh1      = (const float*)d_in[8];   // [2][500][125]
  const float* b1        = (const float*)d_in[9];   // [2][500]
  const float* fc1_W     = (const float*)d_in[10];  // [100][500]
  const float* fc1_b     = (const float*)d_in[11];  // [100]
  const float* fc2_W     = (const float*)d_in[12];  // [1][100]
  const float* fc2_b     = (const float*)d_in[13];  // [1]

  // ---- workspace carve-up (256B aligned) ----
  char* ws = (char*)d_ws;
  size_t off = 0;
  auto carve = [&](size_t bytes) -> char* {
    char* p = ws + off;
    off += (bytes + 255) & ~(size_t)255;
    return p;
  };
  _Float16* x16    = (_Float16*)carve((size_t)1024 * 352 * 2);
  _Float16* W0f16  = (_Float16*)carve((size_t)512 * 352 * 2);
  _Float16* W0b16  = (_Float16*)carve((size_t)512 * 352 * 2);
  _Float16* W1f16  = (_Float16*)carve((size_t)512 * 256 * 2);
  _Float16* W1b16  = (_Float16*)carve((size_t)512 * 256 * 2);
  _Float16* WheadT = (_Float16*)carve((size_t)128 * 256 * 2);
  _Float16* WchildT= (_Float16*)carve((size_t)128 * 256 * 2);
  float* whhT0f    = (float*)carve((size_t)128 * 512 * 4);
  float* whhT0b    = (float*)carve((size_t)128 * 512 * 4);
  float* whhT1f    = (float*)carve((size_t)128 * 512 * 4);
  float* whhT1b    = (float*)carve((size_t)128 * 512 * 4);
  float* bias0f    = (float*)carve(512 * 4);
  float* bias0b    = (float*)carve(512 * 4);
  float* bias1f    = (float*)carve(512 * 4);
  float* bias1b    = (float*)carve(512 * 4);
  float* GpreF     = (float*)carve((size_t)1024 * 512 * 4);
  float* GpreB     = (float*)carve((size_t)1024 * 512 * 4);
  _Float16* h0cat  = (_Float16*)carve((size_t)1024 * 256 * 2);
  _Float16* h1cat  = (_Float16*)carve((size_t)1024 * 256 * 2);
  float* Aproj     = (float*)carve((size_t)1024 * 128 * 4);
  float* Bproj     = (float*)carve((size_t)1024 * 128 * 4);
  float* colloss   = (float*)carve((size_t)1024 * 4);

  float* loss_out = (float*)d_out;
  float* S        = (float*)d_out + 1;   // [1024][1024] row-major (head i, child j)

  // ---- prep / marshalling ----
  build_x_k<<<1024, 352, 0, stream>>>(word_vecs, pos_idx, pos_emb, x16);

  pack_f16_k<<<704, 256, 0, stream>>>(Wih0,             W0f16, 500, 325, 325, 0, 512, 352);
  pack_f16_k<<<704, 256, 0, stream>>>(Wih0 + 500 * 325, W0b16, 500, 325, 325, 0, 512, 352);
  pack_f16_k<<<512, 256, 0, stream>>>(Wih1,             W1f16, 500, 250, 250, 0, 512, 256);
  pack_f16_k<<<512, 256, 0, stream>>>(Wih1 + 500 * 250, W1b16, 500, 250, 250, 0, 512, 256);
  pack_f16_k<<<128, 256, 0, stream>>>(fc1_W, WheadT,  100, 250, 500, 0,   128, 256);
  pack_f16_k<<<128, 256, 0, stream>>>(fc1_W, WchildT, 100, 250, 500, 250, 128, 256);

  pack_whhT_k<<<256, 256, 0, stream>>>(Whh0,             whhT0f);
  pack_whhT_k<<<256, 256, 0, stream>>>(Whh0 + 500 * 125, whhT0b);
  pack_whhT_k<<<256, 256, 0, stream>>>(Whh1,             whhT1f);
  pack_whhT_k<<<256, 256, 0, stream>>>(Whh1 + 500 * 125, whhT1b);

  pack_f32_k<<<2, 256, 0, stream>>>(b0,       bias0f, 1, 500, 500, 0, 1, 512);
  pack_f32_k<<<2, 256, 0, stream>>>(b0 + 500, bias0b, 1, 500, 500, 0, 1, 512);
  pack_f32_k<<<2, 256, 0, stream>>>(b1,       bias1f, 1, 500, 500, 0, 1, 512);
  pack_f32_k<<<2, 256, 0, stream>>>(b1 + 500, bias1b, 1, 500, 500, 0, 1, 512);

  zero_u32_k<<<512, 256, 0, stream>>>((uint32_t*)h0cat, 1024 * 256 / 2);
  zero_u32_k<<<512, 256, 0, stream>>>((uint32_t*)h1cat, 1024 * 256 / 2);

  // ---- layer 0: input projections (WMMA) + recurrence ----
  wmma_gemm_nt<<<dim3(64, 8), 128, 0, stream>>>(x16, W0f16, bias0f, GpreF, 1024, 512, 352, 512);
  wmma_gemm_nt<<<dim3(64, 8), 128, 0, stream>>>(x16, W0b16, bias0b, GpreB, 1024, 512, 352, 512);
  bilstm_recur_k<<<2, 512, 0, stream>>>(GpreF, GpreB, whhT0f, whhT0b, h0cat, N);

  // ---- layer 1 ----
  wmma_gemm_nt<<<dim3(64, 8), 128, 0, stream>>>(h0cat, W1f16, bias1f, GpreF, 1024, 512, 256, 512);
  wmma_gemm_nt<<<dim3(64, 8), 128, 0, stream>>>(h0cat, W1b16, bias1b, GpreB, 1024, 512, 256, 512);
  bilstm_recur_k<<<2, 512, 0, stream>>>(GpreF, GpreB, whhT1f, whhT1b, h1cat, N);

  // ---- head / child projections (WMMA) ----
  wmma_gemm_nt<<<dim3(64, 2), 128, 0, stream>>>(h1cat, WheadT,  nullptr, Aproj, 1024, 128, 256, 128);
  wmma_gemm_nt<<<dim3(64, 2), 128, 0, stream>>>(h1cat, WchildT, nullptr, Bproj, 1024, 128, 256, 128);

  // ---- pairwise scores + masked log-softmax loss ----
  pair_scores_k<<<dim3(32, 32), 256, 0, stream>>>(Aproj, Bproj, fc1_b, fc2_W, fc2_b, S, N);
  col_loss_k<<<1024, 256, 0, stream>>>(S, parents, colloss, N);
  reduce_loss_k<<<1, 256, 0, stream>>>(colloss, loss_out, N);
}
